// EBencoder_14869176779098
// MI455X (gfx1250) — matrix operand order repaired and checked by reading, not compiled, not run on hardware
//
#include <hip/hip_runtime.h>
#include <stdint.h>

typedef unsigned short u16;
typedef __attribute__((ext_vector_type(16))) __bf16 v16bf;
typedef __attribute__((ext_vector_type(8)))  float  v8f;
typedef __attribute__((ext_vector_type(4)))  int    v4i_t;

#define NTOK 2048
#define DDIM 1024
#define IDIM 4096
#define ODIM 65536
#define LDSS 40   // bf16 LDS tile row stride in u16 (80 B, multiple of 16 B)
#define LDSW 36   // fp32 LDS tile row stride in floats (144 B, multiple of 16 B)

typedef __attribute__((address_space(3))) void  lds_void;
typedef __attribute__((address_space(1))) v4i_t gas1_v4i;   // global int4
typedef __attribute__((address_space(3))) v4i_t las3_v4i;   // LDS int4

// round-to-nearest bf16 truncation: bias fp32 bits, keep high 16
__device__ __forceinline__ u16 f2bf(float f) {
  return (u16)((__float_as_uint(f) + 0x8000u) >> 16);
}
__device__ __forceinline__ float bf2f(u16 h) {
  return __uint_as_float(((uint32_t)h) << 16);
}
// pack high halves of two rounded fp32 bit patterns: {b[31:16], a[31:16]}
// single v_perm_b32: sel bytes {7,6,3,2} of {s0=b, s1=a}
__device__ __forceinline__ uint32_t pack2bf(uint32_t ua, uint32_t ub) {
  return __builtin_amdgcn_perm(ub, ua, 0x07060302u);
}

union Frag { v16bf v; uint4 q[2]; };

__device__ __forceinline__ v8f zero8() {
  v8f z = {0.f, 0.f, 0.f, 0.f, 0.f, 0.f, 0.f, 0.f};
  return z;
}

// ---- 16-byte async global->LDS copy (ASYNCcnt-tracked on gfx1250) ----
__device__ __forceinline__ void async_cp16(void* dst, const void* src) {
#if defined(__AMDGCN__) && __has_builtin(__builtin_amdgcn_global_load_async_to_lds_b128)
  __builtin_amdgcn_global_load_async_to_lds_b128(
      (gas1_v4i*)(unsigned long long)src,
      (las3_v4i*)(lds_void*)dst, 0, 0);
#elif defined(__gfx1250__)
  unsigned int loff = (unsigned int)(unsigned long long)(lds_void*)dst;
  unsigned long long ga = (unsigned long long)src;
  asm volatile("global_load_async_to_lds_b128 %0, %1, off"
               :: "v"(loff), "v"(ga) : "memory");
#else
  *(uint4*)dst = *(const uint4*)src;
#endif
}

__device__ __forceinline__ void wait_async0() {
#if defined(__AMDGCN__) && __has_builtin(__builtin_amdgcn_s_wait_asynccnt)
  __builtin_amdgcn_s_wait_asynccnt(0);
#elif defined(__gfx1250__)
  asm volatile("s_wait_asynccnt 0x0" ::: "memory");
#endif
}

// ---- async-stage a 128x32 bf16 tile (fragment-ready padded layout) ----
__device__ __forceinline__ void stage_a_async(u16* sm, const u16* __restrict__ src,
                                              int ld, int m0, int k0, int t) {
  const u16* base = src + (size_t)m0 * ld + k0;
#pragma unroll
  for (int i = 0; i < 2; ++i) {
    int c = t + i * 256;
    int row = c >> 2, sub = c & 3;
    async_cp16(&sm[row * LDSS + sub * 8], base + (size_t)row * ld + sub * 8);
  }
}

// ---- async-stage a 128x32 fp32 weight tile (raw fp32; convert at frag load) ----
__device__ __forceinline__ void stage_w_async(float* sm, const float* __restrict__ src,
                                              int ld, int n0, int k0, int t) {
  const float* base = src + (size_t)n0 * ld + k0;
#pragma unroll
  for (int i = 0; i < 4; ++i) {
    int c = t + i * 256;
    int row = c >> 3, sub = c & 7;
    async_cp16(&sm[row * LDSW + sub * 4], base + (size_t)row * ld + sub * 4);
  }
}

// ---- A fragment: lane r holds row m, K=hi*16..+15 ----
__device__ __forceinline__ Frag load_a(const u16* sm, int mrow, int lane) {
  int r = lane & 15, hi = lane >> 4;
  const u16* p = sm + (mrow + r) * LDSS + hi * 16;
  Frag f;
  f.q[0] = *(const uint4*)p;
  f.q[1] = *(const uint4*)(p + 8);
  return f;
}

// ---- B fragment from fp32 LDS tile: lane r = column n, K=hi*16..+15, cvt->bf16 ----
__device__ __forceinline__ Frag load_b32(const float* sm, int nrow, int lane) {
  int r = lane & 15, hi = lane >> 4;
  const uint4* p = (const uint4*)(sm + (nrow + r) * LDSW + hi * 16);
  uint4 q0 = p[0], q1 = p[1], q2 = p[2], q3 = p[3];
  q0.x += 0x8000u; q0.y += 0x8000u; q0.z += 0x8000u; q0.w += 0x8000u;
  q1.x += 0x8000u; q1.y += 0x8000u; q1.z += 0x8000u; q1.w += 0x8000u;
  q2.x += 0x8000u; q2.y += 0x8000u; q2.z += 0x8000u; q2.w += 0x8000u;
  q3.x += 0x8000u; q3.y += 0x8000u; q3.z += 0x8000u; q3.w += 0x8000u;
  Frag f;
  f.q[0].x = pack2bf(q0.x, q0.y);
  f.q[0].y = pack2bf(q0.z, q0.w);
  f.q[0].z = pack2bf(q1.x, q1.y);
  f.q[0].w = pack2bf(q1.z, q1.w);
  f.q[1].x = pack2bf(q2.x, q2.y);
  f.q[1].y = pack2bf(q2.z, q2.w);
  f.q[1].z = pack2bf(q3.x, q3.y);
  f.q[1].w = pack2bf(q3.z, q3.w);
  return f;
}

// ===================== kernel 1: gather + router =====================
__global__ void __launch_bounds__(128) k_gather_router(
    const int* __restrict__ idx, const float* __restrict__ emb,
    const float* __restrict__ gw, u16* __restrict__ xbf,
    float* __restrict__ wstat, float* __restrict__ wdyn)
{
  __shared__ float red[128][8];
  int n = blockIdx.x;
  int t = threadIdx.x;
  int tok = idx[n];
  const float* er = emb + (size_t)tok * DDIM;
  float acc[8];
#pragma unroll
  for (int e = 0; e < 8; ++e) acc[e] = 0.f;
#pragma unroll
  for (int j = 0; j < 8; ++j) {
    int d = t + j * 128;
    float v = er[d];
    xbf[(size_t)n * DDIM + d] = f2bf(v);
#pragma unroll
    for (int e = 0; e < 8; ++e) acc[e] += v * gw[e * DDIM + d];
  }
#pragma unroll
  for (int e = 0; e < 8; ++e) red[t][e] = acc[e];
  __syncthreads();
  for (int s = 64; s > 0; s >>= 1) {
    if (t < s) {
#pragma unroll
      for (int e = 0; e < 8; ++e) red[t][e] += red[t + s][e];
    }
    __syncthreads();
  }
  if (t == 0) {
    float l[8], m = -1e30f;
#pragma unroll
    for (int e = 0; e < 8; ++e) { l[e] = red[0][e]; m = fmaxf(m, l[e]); }
    float s = 0.f, p[8];
#pragma unroll
    for (int e = 0; e < 8; ++e) { p[e] = __expf(l[e] - m); s += p[e]; }
    float inv = __builtin_amdgcn_rcpf(s);
#pragma unroll
    for (int e = 0; e < 8; ++e) p[e] *= inv;
    int i1 = 0;
    for (int e = 1; e < 8; ++e) if (p[e] > p[i1]) i1 = e;
    int i2 = (i1 == 0) ? 1 : 0;
    for (int e = 0; e < 8; ++e) if (e != i1 && p[e] > p[i2]) i2 = e;
    float w1 = p[i1], w2 = p[i2];
    float rt = __builtin_amdgcn_rcpf(w1 + w2);
    w1 *= rt; w2 *= rt;
    float ws = 0.f, wd = 0.f;
    if (i1 < 2) ws += w1; else wd += w1;
    if (i2 < 2) ws += w2; else wd += w2;
    wstat[n] = ws;
    wdyn[n] = wd;
  }
}

// ===================== kernel 2: fused gate+up GEMM + SwiGLU (single-buffer async) ===
// grid: (x = M blocks, y = N blocks) so weight tiles are L2-shared across M
__global__ void __launch_bounds__(256) k_gateup(
    const u16* __restrict__ xbf, const float* __restrict__ wg,
    const float* __restrict__ wu, u16* __restrict__ h)
{
  __shared__ __align__(16) u16   As[128 * LDSS];
  __shared__ __align__(16) float Bg[128 * LDSW];
  __shared__ __align__(16) float Bu[128 * LDSW];
  int t = threadIdx.x;
  int lane = t & 31, wv = t >> 5;
  int wm = wv & 1, wn = wv >> 1;
  int m0 = blockIdx.x * 128, n0 = blockIdx.y * 128;
  v8f ag[4][2], au[4][2];
#pragma unroll
  for (int a = 0; a < 4; ++a)
#pragma unroll
    for (int b = 0; b < 2; ++b) { ag[a][b] = zero8(); au[a][b] = zero8(); }

  for (int k0 = 0; k0 < DDIM; k0 += 32) {
    stage_a_async(As, xbf, DDIM, m0, k0, t);
    stage_w_async(Bg, wg, DDIM, n0, k0, t);
    stage_w_async(Bu, wu, DDIM, n0, k0, t);
    wait_async0();
    __syncthreads();
    Frag fa[4];
#pragma unroll
    for (int mm = 0; mm < 4; ++mm) fa[mm] = load_a(As, wm * 64 + mm * 16, lane);
#pragma unroll
    for (int nn = 0; nn < 2; ++nn) {
      Frag fbg = load_b32(Bg, wn * 32 + nn * 16, lane);
      Frag fbu = load_b32(Bu, wn * 32 + nn * 16, lane);
#pragma unroll
      for (int mm = 0; mm < 4; ++mm) {
        ag[mm][nn] = __builtin_amdgcn_wmma_f32_16x16x32_bf16(
            false, fa[mm].v, false, fbg.v, (short)0, ag[mm][nn], false, false);
        au[mm][nn] = __builtin_amdgcn_wmma_f32_16x16x32_bf16(
            false, fa[mm].v, false, fbu.v, (short)0, au[mm][nn], false, false);
      }
    }
    __syncthreads();
  }
  int r = lane & 15, hi = lane >> 4;
#pragma unroll
  for (int nn = 0; nn < 2; ++nn) {
    int col = n0 + wn * 32 + nn * 16 + r;
#pragma unroll
    for (int mm = 0; mm < 4; ++mm) {
      int rowb = m0 + wm * 64 + mm * 16 + hi * 8;
#pragma unroll
      for (int j = 0; j < 8; ++j) {
        float g = ag[mm][nn][j];
        float u = au[mm][nn][j];
        float sv = g * __builtin_amdgcn_rcpf(1.f + __expf(-g));   // SiLU, v_rcp_f32
        h[(size_t)(rowb + j) * IDIM + col] = f2bf(sv * u);
      }
    }
  }
}

// ===================== kernel 3: down GEMM + expert combine (double-buffered) =====
__global__ void __launch_bounds__(256) k_down(
    const u16* __restrict__ h, const float* __restrict__ wd,
    const u16* __restrict__ xbf, const float* __restrict__ wstat,
    const float* __restrict__ wdyn, u16* __restrict__ xo)
{
  __shared__ __align__(16) u16   As[2][128 * LDSS];
  __shared__ __align__(16) float Bs[2][128 * LDSW];
  int t = threadIdx.x;
  int lane = t & 31, wv = t >> 5;
  int wm = wv & 1, wn = wv >> 1;
  int m0 = blockIdx.x * 128, n0 = blockIdx.y * 128;
  v8f acc[4][2];
#pragma unroll
  for (int a = 0; a < 4; ++a)
#pragma unroll
    for (int b = 0; b < 2; ++b) acc[a][b] = zero8();

  const int KSTEPS = IDIM / 32;
  stage_a_async(As[0], h, IDIM, m0, 0, t);
  stage_w_async(Bs[0], wd, IDIM, n0, 0, t);
  for (int kidx = 0; kidx < KSTEPS; ++kidx) {
    int buf = kidx & 1;
    wait_async0();
    __syncthreads();
    if (kidx + 1 < KSTEPS) {
      stage_a_async(As[buf ^ 1], h, IDIM, m0, (kidx + 1) * 32, t);
      stage_w_async(Bs[buf ^ 1], wd, IDIM, n0, (kidx + 1) * 32, t);
    }
    Frag fa[4];
#pragma unroll
    for (int mm = 0; mm < 4; ++mm) fa[mm] = load_a(As[buf], wm * 64 + mm * 16, lane);
#pragma unroll
    for (int nn = 0; nn < 2; ++nn) {
      Frag fb = load_b32(Bs[buf], wn * 32 + nn * 16, lane);
#pragma unroll
      for (int mm = 0; mm < 4; ++mm) {
        acc[mm][nn] = __builtin_amdgcn_wmma_f32_16x16x32_bf16(
            false, fa[mm].v, false, fb.v, (short)0, acc[mm][nn], false, false);
      }
    }
  }
  int r = lane & 15, hi = lane >> 4;
#pragma unroll
  for (int nn = 0; nn < 2; ++nn) {
    int col = n0 + wn * 32 + nn * 16 + r;
#pragma unroll
    for (int mm = 0; mm < 4; ++mm) {
      int rowb = m0 + wm * 64 + mm * 16 + hi * 8;
#pragma unroll
      for (int j = 0; j < 8; ++j) {
        int row = rowb + j;
        float ws = wstat[row], wdn = wdyn[row];
        float xv = bf2f(xbf[(size_t)row * DDIM + col]);
        xo[(size_t)row * DDIM + col] = f2bf(ws * xv + wdn * acc[mm][nn][j]);
      }
    }
  }
}

// ===================== kernel 4: final transform GEMM + bias (double-buffered) =====
__global__ void __launch_bounds__(256) k_transform(
    const u16* __restrict__ xo, const float* __restrict__ wt,
    const float* __restrict__ bias, float* __restrict__ out)
{
  __shared__ __align__(16) u16   As[2][128 * LDSS];
  __shared__ __align__(16) float Bs[2][128 * LDSW];
  int t = threadIdx.x;
  int lane = t & 31, wv = t >> 5;
  int wm = wv & 1, wn = wv >> 1;
  int m0 = blockIdx.x * 128, n0 = blockIdx.y * 128;
  v8f acc[4][2];
#pragma unroll
  for (int a = 0; a < 4; ++a)
#pragma unroll
    for (int b = 0; b < 2; ++b) acc[a][b] = zero8();

  const int KSTEPS = DDIM / 32;
  stage_a_async(As[0], xo, DDIM, m0, 0, t);
  stage_w_async(Bs[0], wt, DDIM, n0, 0, t);
  for (int kidx = 0; kidx < KSTEPS; ++kidx) {
    int buf = kidx & 1;
    wait_async0();
    __syncthreads();
    if (kidx + 1 < KSTEPS) {
      stage_a_async(As[buf ^ 1], xo, DDIM, m0, (kidx + 1) * 32, t);
      stage_w_async(Bs[buf ^ 1], wt, DDIM, n0, (kidx + 1) * 32, t);
    }
    Frag fa[4];
#pragma unroll
    for (int mm = 0; mm < 4; ++mm) fa[mm] = load_a(As[buf], wm * 64 + mm * 16, lane);
#pragma unroll
    for (int nn = 0; nn < 2; ++nn) {
      Frag fb = load_b32(Bs[buf], wn * 32 + nn * 16, lane);
#pragma unroll
      for (int mm = 0; mm < 4; ++mm) {
        acc[mm][nn] = __builtin_amdgcn_wmma_f32_16x16x32_bf16(
            false, fa[mm].v, false, fb.v, (short)0, acc[mm][nn], false, false);
      }
    }
  }
  int r = lane & 15, hi = lane >> 4;
#pragma unroll
  for (int nn = 0; nn < 2; ++nn) {
    int col = n0 + wn * 32 + nn * 16 + r;
    float bv = bias[col];
#pragma unroll
    for (int mm = 0; mm < 4; ++mm) {
      int rowb = m0 + wm * 64 + mm * 16 + hi * 8;
#pragma unroll
      for (int j = 0; j < 8; ++j) {
        out[(size_t)(rowb + j) * ODIM + col] = acc[mm][nn][j] + bv;
      }
    }
  }
}

extern "C" void kernel_launch(void* const* d_in, const int* in_sizes, int n_in,
                              void* d_out, int out_size, void* d_ws, size_t ws_size,
                              hipStream_t stream) {
  const int*   indices = (const int*)d_in[0];
  const float* emb     = (const float*)d_in[1];
  const float* gate_w  = (const float*)d_in[2];
  const float* wgate   = (const float*)d_in[3];
  const float* wup     = (const float*)d_in[4];
  const float* wdown   = (const float*)d_in[5];
  const float* wtrans  = (const float*)d_in[6];
  const float* btrans  = (const float*)d_in[7];
  float* out = (float*)d_out;

  char* ws = (char*)d_ws;
  u16*   xbf   = (u16*)(ws);                                 // 4 MB  bf16 x
  u16*   hbuf  = (u16*)(ws + (size_t)(4u << 20));            // 16 MB bf16 h
  u16*   xo    = (u16*)(ws + (size_t)(20u << 20));           // 4 MB  bf16 combined
  float* wstat = (float*)(ws + (size_t)(24u << 20));         // 8 KB
  float* wdyn  = (float*)(ws + (size_t)(24u << 20) + 8192);  // 8 KB

  // grid x = M (fast-varying) so blocks sharing a weight tile are adjacent -> L2 reuse
  k_gather_router<<<NTOK, 128, 0, stream>>>(indices, emb, gate_w, xbf, wstat, wdyn);
  k_gateup<<<dim3(NTOK / 128, IDIM / 128), 256, 0, stream>>>(xbf, wgate, wup, hbuf);
  k_down<<<dim3(NTOK / 128, DDIM / 128), 256, 0, stream>>>(hbuf, wdown, xbf, wstat, wdyn, xo);
  k_transform<<<dim3(NTOK / 128, ODIM / 128), 256, 0, stream>>>(xo, wtrans, btrans, out);
}
